// NextTaskGAT_26938034880990
// MI455X (gfx1250) — matrix operand order repaired and checked by reading, not compiled, not run on hardware
//
#include <hip/hip_runtime.h>
#include <hip/hip_bf16.h>
#include <math.h>

#define DEVFN __device__ __forceinline__

constexpr int   N_   = 50000;
constexpr int   E_   = 800000;
constexpr int   FIN  = 128;
constexpr int   HD   = 64;
constexpr int   H_   = 4;
constexpr int   G_   = 512;
constexpr int   OUT_ = 64;
constexpr int   HC   = HD * H_;          // 256
constexpr int   ETOT = E_ + N_;          // self loops appended
constexpr float LN_EPS = 1e-5f;

typedef __attribute__((ext_vector_type(16))) __bf16 v16bf;
typedef __attribute__((ext_vector_type(8)))  float  v8f;

union Frag { v16bf v; uint4 q[2]; };

DEVFN unsigned short f2bf(float f) {
    unsigned u = __float_as_uint(f);
    u += 0x7FFFu + ((u >> 16) & 1u);        // round-to-nearest-even
    return (unsigned short)(u >> 16);
}
DEVFN unsigned f2ord(float f) {             // order-preserving float->uint
    unsigned u = __float_as_uint(f);
    return (u & 0x80000000u) ? ~u : (u | 0x80000000u);
}
DEVFN float ord2f(unsigned o) {
    unsigned u = (o & 0x80000000u) ? (o & 0x7FFFFFFFu) : ~o;
    return __uint_as_float(u);
}
DEVFN float gelu_f(float x) { return 0.5f * x * (1.0f + erff(x * 0.70710678118654752f)); }

DEVFN void edge_sd(const int* __restrict__ ei, int e, int& s, int& d) {
    if (e < E_) { s = ei[e]; d = ei[E_ + e]; }
    else        { s = e - E_; d = s; }
}

// ---------------------------------------------------------------------------
// Pack f32 weight matrix W[K,Nout] into CDNA5 WMMA B-fragment tile layout:
// tile (ntile,ktile) of 32K x 16N -> 512 bf16, index = lane*16 + elem,
// lane<16:  col=lane,    K = ktile*32 + 0..15  (elems 0..15)
// lane>=16: col=lane-16, K = ktile*32 + 16..31
// ---------------------------------------------------------------------------
__global__ void k_pack(const float* __restrict__ W, unsigned short* __restrict__ P,
                       int K, int Nout) {
    int t = blockIdx.x * blockDim.x + threadIdx.x;
    if (t >= K * Nout) return;
    int tile = t >> 9, w = t & 511;
    int lane = w >> 4, e = w & 15;
    int Ktiles = K >> 5;
    int ntile = tile / Ktiles, ktile = tile % Ktiles;
    int n = ntile * 16 + (lane & 15);
    int k = ktile * 32 + ((lane >= 16) ? 16 : 0) + e;
    P[t] = f2bf(W[(size_t)k * Nout + n]);
}

// concat(x, group_table[gid]) -> bf16 [N,192]
__global__ void k_hcat(const float* __restrict__ x, const float* __restrict__ gt,
                       const int* __restrict__ gid, unsigned short* __restrict__ outb) {
    int t = blockIdx.x * blockDim.x + threadIdx.x;
    if (t >= N_ * 192) return;
    int n = t / 192, c = t % 192;
    float v = (c < FIN) ? x[(size_t)n * FIN + c] : gt[(size_t)gid[n] * HD + (c - FIN)];
    outb[t] = f2bf(v);
}

// ---------------------------------------------------------------------------
// WMMA GEMM: C[M,Nout] (f32) = A[M,K] (bf16 row-major, stride lda) * Bpack
// block = 128 threads (4 waves). Wave -> 16-row strip, 64 columns (4 WMMAs/K-step).
// A fragment loaded in contiguous k order (k = 16*(lane/16) + elem); B fragments are pre-packed contiguous.
// ---------------------------------------------------------------------------
__global__ void k_gemm_bf16(const unsigned short* __restrict__ A, int lda, int K,
                            const unsigned short* __restrict__ Bp, int Nout,
                            float* __restrict__ C, int M) {
    int wave = threadIdx.x >> 5, lane = threadIdx.x & 31;
    int mtile = blockIdx.x * 4 + wave;
    if (mtile * 16 >= M) return;
    int Ktiles = K >> 5;
    int n0tile = blockIdx.y * 4;
    const unsigned short* Arow = A + (size_t)(mtile * 16 + (lane & 15)) * lda;
    int koff = (lane < 16) ? 0 : 16;

    v8f acc[4] = {};
    for (int kt = 0; kt < Ktiles; ++kt) {
        Frag a;
        int k0 = kt * 32 + koff;
        a.q[0] = *reinterpret_cast<const uint4*>(Arow + k0);
        a.q[1] = *reinterpret_cast<const uint4*>(Arow + k0 + 8);
#pragma unroll
        for (int j = 0; j < 4; ++j) {
            const unsigned short* bp =
                Bp + ((size_t)(n0tile + j) * Ktiles + kt) * 512 + lane * 16;
            Frag b;
            b.q[0] = *reinterpret_cast<const uint4*>(bp);
            b.q[1] = *reinterpret_cast<const uint4*>(bp + 8);
            acc[j] = __builtin_amdgcn_wmma_f32_16x16x32_bf16(
                false, a.v, false, b.v, (short)0, acc[j], false, false);
        }
    }
    // C/D layout: VGPR r -> M = r (lanes 0-15) / 8+r (lanes 16-31), N = lane&15
    int rbase = mtile * 16 + ((lane < 16) ? 0 : 8);
    int cbase = n0tile * 16 + (lane & 15);
#pragma unroll
    for (int r = 0; r < 8; ++r) {
        float* cp = C + (size_t)(rbase + r) * Nout + cbase;
        cp[0]  = acc[0][r];
        cp[16] = acc[1][r];
        cp[32] = acc[2][r];
        cp[48] = acc[3][r];
    }
}

// LN + GELU over 64 channels, wave per node (wave32: 2 ch/lane) -> bf16 out
__global__ void k_ln_gelu_64(const float* __restrict__ X, const float* __restrict__ bias,
                             const float* __restrict__ g, const float* __restrict__ b,
                             unsigned short* __restrict__ outb) {
    int wave = threadIdx.x >> 5, lane = threadIdx.x & 31;
    int n = blockIdx.x * 8 + wave;
    if (n >= N_) return;
    const float* xr = X + (size_t)n * 64;
    float v0 = xr[lane] + bias[lane];
    float v1 = xr[lane + 32] + bias[lane + 32];
    float s = v0 + v1;
    for (int m = 16; m; m >>= 1) s += __shfl_xor(s, m);
    float mu = s * (1.0f / 64.0f);
    float d0 = v0 - mu, d1 = v1 - mu;
    float q = d0 * d0 + d1 * d1;
    for (int m = 16; m; m >>= 1) q += __shfl_xor(q, m);
    float rs = rsqrtf(q * (1.0f / 64.0f) + LN_EPS);
    outb[(size_t)n * 64 + lane]      = f2bf(gelu_f(d0 * rs * g[lane] + b[lane]));
    outb[(size_t)n * 64 + lane + 32] = f2bf(gelu_f(d1 * rs * g[lane + 32] + b[lane + 32]));
}

// LN + GELU over 256 channels (8 ch/lane), writes f32 and bf16 copies
__global__ void k_ln_gelu_256(const float* __restrict__ X, const float* __restrict__ bias,
                              const float* __restrict__ g, const float* __restrict__ b,
                              float* __restrict__ outf, unsigned short* __restrict__ outb) {
    int wave = threadIdx.x >> 5, lane = threadIdx.x & 31;
    int n = blockIdx.x * 8 + wave;
    if (n >= N_) return;
    const float* xr = X + (size_t)n * HC;
    float v[8], s = 0.0f;
#pragma unroll
    for (int i = 0; i < 8; ++i) { v[i] = xr[lane + 32 * i] + bias[lane + 32 * i]; s += v[i]; }
    for (int m = 16; m; m >>= 1) s += __shfl_xor(s, m);
    float mu = s * (1.0f / 256.0f), q = 0.0f;
#pragma unroll
    for (int i = 0; i < 8; ++i) { float d = v[i] - mu; q += d * d; }
    for (int m = 16; m; m >>= 1) q += __shfl_xor(q, m);
    float rs = rsqrtf(q * (1.0f / 256.0f) + LN_EPS);
#pragma unroll
    for (int i = 0; i < 8; ++i) {
        int c = lane + 32 * i;
        float o = gelu_f((v[i] - mu) * rs * g[c] + b[c]);
        outf[(size_t)n * HC + c] = o;
        outb[(size_t)n * HC + c] = f2bf(o);
    }
}

// LN + GELU + residual over 256 channels, f32 out only
__global__ void k_ln_gelu_res_256(const float* __restrict__ X, const float* __restrict__ bias,
                                  const float* __restrict__ g, const float* __restrict__ b,
                                  const float* __restrict__ idn, float* __restrict__ outf) {
    int wave = threadIdx.x >> 5, lane = threadIdx.x & 31;
    int n = blockIdx.x * 8 + wave;
    if (n >= N_) return;
    const float* xr = X + (size_t)n * HC;
    float v[8], s = 0.0f;
#pragma unroll
    for (int i = 0; i < 8; ++i) { v[i] = xr[lane + 32 * i] + bias[lane + 32 * i]; s += v[i]; }
    for (int m = 16; m; m >>= 1) s += __shfl_xor(s, m);
    float mu = s * (1.0f / 256.0f), q = 0.0f;
#pragma unroll
    for (int i = 0; i < 8; ++i) { float d = v[i] - mu; q += d * d; }
    for (int m = 16; m; m >>= 1) q += __shfl_xor(q, m);
    float rs = rsqrtf(q * (1.0f / 256.0f) + LN_EPS);
#pragma unroll
    for (int i = 0; i < 8; ++i) {
        int c = lane + 32 * i;
        float o = gelu_f((v[i] - mu) * rs * g[c] + b[c]);
        outf[(size_t)n * HC + c] = o + idn[(size_t)n * HC + c];
    }
}

// per-(node,head) attention pre-scores: als = <xl_h, a_src_h>, ald = <xl_h, a_dst_h>
__global__ void k_attn(const float* __restrict__ xl, const float* __restrict__ as_,
                       const float* __restrict__ ad_, float* __restrict__ als,
                       float* __restrict__ ald) {
    int t = blockIdx.x * blockDim.x + threadIdx.x;
    if (t >= N_ * H_) return;
    int n = t >> 2, h = t & 3;
    const float* xr = xl + (size_t)n * HC + h * HD;
    const float* a1 = as_ + h * HD;
    const float* a2 = ad_ + h * HD;
    float s1 = 0.0f, s2 = 0.0f;
#pragma unroll 8
    for (int i = 0; i < HD; ++i) { float v = xr[i]; s1 += v * a1[i]; s2 += v * a2[i]; }
    als[t] = s1; ald[t] = s2;
}

// edge pass 1: e = leaky_relu(als[src]+ald[dst]); store; atomicMax into m[dst]
__global__ void k_edge_max(const int* __restrict__ ei, const float* __restrict__ als,
                           const float* __restrict__ ald, float* __restrict__ ex,
                           unsigned* __restrict__ m) {
    int e = blockIdx.x * blockDim.x + threadIdx.x;
    if (e >= ETOT) return;
    int s, d; edge_sd(ei, e, s, d);
    float4 a = reinterpret_cast<const float4*>(als)[s];
    float4 b = reinterpret_cast<const float4*>(ald)[d];
    float4 v;
    float t;
    t = a.x + b.x; v.x = t > 0.0f ? t : 0.2f * t;
    t = a.y + b.y; v.y = t > 0.0f ? t : 0.2f * t;
    t = a.z + b.z; v.z = t > 0.0f ? t : 0.2f * t;
    t = a.w + b.w; v.w = t > 0.0f ? t : 0.2f * t;
    reinterpret_cast<float4*>(ex)[e] = v;
    atomicMax(&m[d * 4 + 0], f2ord(v.x));
    atomicMax(&m[d * 4 + 1], f2ord(v.y));
    atomicMax(&m[d * 4 + 2], f2ord(v.z));
    atomicMax(&m[d * 4 + 3], f2ord(v.w));
}

// edge pass 2: ex = exp(e - m[dst]); atomicAdd into ssum[dst]
__global__ void k_edge_exp(const int* __restrict__ ei, const unsigned* __restrict__ m,
                           float* __restrict__ ex, float* __restrict__ ssum) {
    int e = blockIdx.x * blockDim.x + threadIdx.x;
    if (e >= ETOT) return;
    int s, d; edge_sd(ei, e, s, d);
    float4 v = reinterpret_cast<const float4*>(ex)[e];
    uint4 mo = reinterpret_cast<const uint4*>(m)[d];
    v.x = expf(v.x - ord2f(mo.x));
    v.y = expf(v.y - ord2f(mo.y));
    v.z = expf(v.z - ord2f(mo.z));
    v.w = expf(v.w - ord2f(mo.w));
    reinterpret_cast<float4*>(ex)[e] = v;
    atomicAdd(&ssum[d * 4 + 0], v.x);
    atomicAdd(&ssum[d * 4 + 1], v.y);
    atomicAdd(&ssum[d * 4 + 2], v.z);
    atomicAdd(&ssum[d * 4 + 3], v.w);
}

// edge pass 3: agg[dst] += xl[src] * alpha ; thread = (edge, 4-channel chunk)
__global__ void k_edge_agg(const int* __restrict__ ei, const float* __restrict__ xl,
                           const float* __restrict__ ex, const float* __restrict__ ssum,
                           float* __restrict__ agg) {
    long long t = (long long)blockIdx.x * blockDim.x + threadIdx.x;
    if (t >= (long long)ETOT * 64) return;
    int e  = (int)(t >> 6);
    int c4 = (int)(t & 63);                 // float4 chunk index, channels c4*4..c4*4+3
    int s, d; edge_sd(ei, e, s, d);
    int h = c4 >> 4;                        // head = (c4*4)/64
    float alpha = ex[(size_t)e * 4 + h] / (ssum[(size_t)d * 4 + h] + 1e-16f);
    float4 v = reinterpret_cast<const float4*>(xl + (size_t)s * HC)[c4];
    float* ap = agg + (size_t)d * HC + c4 * 4;
    atomicAdd(ap + 0, v.x * alpha);
    atomicAdd(ap + 1, v.y * alpha);
    atomicAdd(ap + 2, v.z * alpha);
    atomicAdd(ap + 3, v.w * alpha);
}

// mean-pool accumulation
__global__ void k_pool(const float* __restrict__ h, const int* __restrict__ batch,
                       float* __restrict__ sums, float* __restrict__ cnt) {
    int t = blockIdx.x * blockDim.x + threadIdx.x;
    if (t >= N_ * HC) return;
    int n = t >> 8, c = t & 255;
    int g = batch[n];
    atomicAdd(&sums[(size_t)g * HC + c], h[t]);
    if (c == 0) atomicAdd(&cnt[g], 1.0f);
}

// MLP head + log_softmax: one block (64 threads) per graph
__global__ void k_head(const float* __restrict__ sums, const float* __restrict__ cnt,
                       const float* __restrict__ fc1W, const float* __restrict__ fc1b,
                       const float* __restrict__ fc2W, const float* __restrict__ fc2b,
                       float* __restrict__ out) {
    __shared__ float pooled[HC];
    __shared__ float z[OUT_];
    __shared__ float red[OUT_];
    int g = blockIdx.x, j = threadIdx.x;
    float invc = 1.0f / fmaxf(cnt[g], 1.0f);
    for (int i = j; i < HC; i += OUT_) pooled[i] = sums[(size_t)g * HC + i] * invc;
    __syncthreads();
    float acc = fc1b[j];
    for (int i = 0; i < HC; ++i) acc += pooled[i] * fc1W[(size_t)i * OUT_ + j];
    z[j] = gelu_f(acc);
    __syncthreads();
    float lg = fc2b[j];
    for (int i = 0; i < OUT_; ++i) lg += z[i] * fc2W[(size_t)i * OUT_ + j];
    red[j] = lg; __syncthreads();
    for (int s = 32; s; s >>= 1) { if (j < s) red[j] = fmaxf(red[j], red[j + s]); __syncthreads(); }
    float mx = red[0]; __syncthreads();
    red[j] = expf(lg - mx); __syncthreads();
    for (int s = 32; s; s >>= 1) { if (j < s) red[j] += red[j + s]; __syncthreads(); }
    out[(size_t)g * OUT_ + j] = lg - mx - logf(red[0]);
}

extern "C" void kernel_launch(void* const* d_in, const int* in_sizes, int n_in,
                              void* d_out, int out_size, void* d_ws, size_t ws_size,
                              hipStream_t stream) {
    const float* x      = (const float*)d_in[0];
    const int*   ei     = (const int*)d_in[1];
    const int*   batch  = (const int*)d_in[2];
    const int*   gid    = (const int*)d_in[3];
    const float* gt     = (const float*)d_in[4];
    const float* inW    = (const float*)d_in[5];
    const float* inb    = (const float*)d_in[6];
    const float* inlng  = (const float*)d_in[7];
    const float* inlnb  = (const float*)d_in[8];
    const float* W0     = (const float*)d_in[9];
    const float* as0    = (const float*)d_in[10];
    const float* ad0    = (const float*)d_in[11];
    const float* b0     = (const float*)d_in[12];
    const float* ln0g   = (const float*)d_in[13];
    const float* ln0b   = (const float*)d_in[14];
    const float* W1     = (const float*)d_in[15];
    const float* as1    = (const float*)d_in[16];
    const float* ad1    = (const float*)d_in[17];
    const float* b1     = (const float*)d_in[18];
    const float* ln1g   = (const float*)d_in[19];
    const float* ln1b   = (const float*)d_in[20];
    const float* fc1W   = (const float*)d_in[21];
    const float* fc1b   = (const float*)d_in[22];
    const float* fc2W   = (const float*)d_in[23];
    const float* fc2b   = (const float*)d_in[24];
    float* out = (float*)d_out;

    char* ws = (char*)d_ws;
    size_t off = 0;
    auto alloc = [&](size_t bytes) -> char* {
        char* p = ws + off;
        off = (off + bytes + 255) & ~(size_t)255;
        return p;
    };
    unsigned short* pIn  = (unsigned short*)alloc((size_t)192 * 64 * 2);
    unsigned short* pW0  = (unsigned short*)alloc((size_t)64 * 256 * 2);
    unsigned short* pW1  = (unsigned short*)alloc((size_t)256 * 256 * 2);
    unsigned short* hcat = (unsigned short*)alloc((size_t)N_ * 192 * 2);
    float*          t0   = (float*)alloc((size_t)N_ * 64 * 4);
    unsigned short* h0b  = (unsigned short*)alloc((size_t)N_ * 64 * 2);
    float*          xl   = (float*)alloc((size_t)N_ * HC * 4);   // also reused as h2
    float*          als  = (float*)alloc((size_t)N_ * H_ * 4);
    float*          ald  = (float*)alloc((size_t)N_ * H_ * 4);
    unsigned*       mbuf = (unsigned*)alloc((size_t)N_ * H_ * 4);
    float*          ssum = (float*)alloc((size_t)N_ * H_ * 4);
    float*          ex   = (float*)alloc((size_t)ETOT * H_ * 4);
    float*          agg  = (float*)alloc((size_t)N_ * HC * 4);
    float*          h1f  = (float*)alloc((size_t)N_ * HC * 4);
    unsigned short* h1b  = (unsigned short*)alloc((size_t)N_ * HC * 2);
    float*          sums = (float*)alloc((size_t)G_ * HC * 4);
    float*          cnt  = (float*)alloc((size_t)G_ * 4);
    (void)ws_size; (void)n_in; (void)in_sizes; (void)out_size;

    const int TB = 256;
    auto blks = [](long long n, int tb) { return (unsigned)((n + tb - 1) / tb); };

    // weight packing
    k_pack<<<blks(192 * 64, TB), TB, 0, stream>>>(inW, pIn, 192, 64);
    k_pack<<<blks(64 * 256, TB), TB, 0, stream>>>(W0, pW0, 64, 256);
    k_pack<<<blks(256 * 256, TB), TB, 0, stream>>>(W1, pW1, 256, 256);

    // input concat -> bf16, projection (WMMA), LN+GELU
    k_hcat<<<blks((long long)N_ * 192, TB), TB, 0, stream>>>(x, gt, gid, hcat);
    {
        dim3 g((N_ / 16 + 3) / 4, 1);
        k_gemm_bf16<<<g, 128, 0, stream>>>(hcat, 192, 192, pIn, 64, t0, N_);
    }
    k_ln_gelu_64<<<blks(N_, 8) , 256, 0, stream>>>(t0, inb, inlng, inlnb, h0b);

    // ---- GAT layer 0 ----
    {
        dim3 g((N_ / 16 + 3) / 4, HC / 64);
        k_gemm_bf16<<<g, 128, 0, stream>>>(h0b, 64, 64, pW0, HC, xl, N_);
    }
    k_attn<<<blks((long long)N_ * H_, TB), TB, 0, stream>>>(xl, as0, ad0, als, ald);
    hipMemsetAsync(mbuf, 0, (size_t)N_ * H_ * 4, stream);
    hipMemsetAsync(ssum, 0, (size_t)N_ * H_ * 4, stream);
    hipMemsetAsync(agg,  0, (size_t)N_ * HC * 4, stream);
    k_edge_max<<<blks(ETOT, TB), TB, 0, stream>>>(ei, als, ald, ex, mbuf);
    k_edge_exp<<<blks(ETOT, TB), TB, 0, stream>>>(ei, mbuf, ex, ssum);
    k_edge_agg<<<blks((long long)ETOT * 64, TB), TB, 0, stream>>>(ei, xl, ex, ssum, agg);
    k_ln_gelu_256<<<blks(N_, 8), 256, 0, stream>>>(agg, b0, ln0g, ln0b, h1f, h1b);

    // ---- GAT layer 1 ----
    {
        dim3 g((N_ / 16 + 3) / 4, HC / 64);
        k_gemm_bf16<<<g, 128, 0, stream>>>(h1b, HC, HC, pW1, HC, xl, N_);
    }
    k_attn<<<blks((long long)N_ * H_, TB), TB, 0, stream>>>(xl, as1, ad1, als, ald);
    hipMemsetAsync(mbuf, 0, (size_t)N_ * H_ * 4, stream);
    hipMemsetAsync(ssum, 0, (size_t)N_ * H_ * 4, stream);
    hipMemsetAsync(agg,  0, (size_t)N_ * HC * 4, stream);
    k_edge_max<<<blks(ETOT, TB), TB, 0, stream>>>(ei, als, ald, ex, mbuf);
    k_edge_exp<<<blks(ETOT, TB), TB, 0, stream>>>(ei, mbuf, ex, ssum);
    k_edge_agg<<<blks((long long)ETOT * 64, TB), TB, 0, stream>>>(ei, xl, ex, ssum, agg);
    // h2 = gelu(LN(agg+b1)) + h1  (written into xl, which is free now)
    k_ln_gelu_res_256<<<blks(N_, 8), 256, 0, stream>>>(agg, b1, ln1g, ln1b, h1f, xl);

    // ---- pooling + head ----
    hipMemsetAsync(sums, 0, (size_t)G_ * HC * 4, stream);
    hipMemsetAsync(cnt,  0, (size_t)G_ * 4, stream);
    k_pool<<<blks((long long)N_ * HC, TB), TB, 0, stream>>>(xl, batch, sums, cnt);
    k_head<<<G_, OUT_, 0, stream>>>(sums, cnt, fc1W, fc1b, fc2W, fc2b, out);
}
